// MultiHeadSelfAttention_88476326298208
// MI455X (gfx1250) — matrix operand order repaired and checked by reading, not compiled, not run on hardware
//
#include <hip/hip_runtime.h>

// MI455X / gfx1250 multi-head self-attention, bf16 WMMA pipeline:
//  1) cvt x -> bf16, transpose+cvt weights -> bf16 [out][in]
//  2) WMMA GEMM (32x128/wave, B-fragments shared across the block's 8 waves):
//     Q,K -> [B,H,T,D] bf16 ; V -> [B,H,D,T] bf16 (all +bias)
//  3) flash-attention per 16-row q strip: WMMA scores, online softmax
//     (row-max via shfl_xor, row-sum via P @ ones WMMA),
//     P staged via LDS (C-layout -> A-layout transpose), WMMA P@V
//  4) WMMA out-projection -> fp32 d_out (+bias)

typedef unsigned short u16;
typedef __attribute__((ext_vector_type(16))) __bf16 v16bf;
typedef __attribute__((ext_vector_type(8)))  float  v8f;

#define Bx 4
#define Tx 2048
#define Cx 1024
#define Hx 16
#define Dx 64
#define Mx (Bx * Tx)   // 8192 rows

__device__ __forceinline__ u16 f2bf(float f) {
    unsigned int u = __float_as_uint(f);
    unsigned int r = u + 0x7FFFu + ((u >> 16) & 1u);   // round-to-nearest-even
    return (u16)(r >> 16);
}

// A-operand fragment (16x32 bf16): lane=row; elems 0..7 at K in [8h,8h+8),
// elems 8..15 at K in [16+8h, 16+8h+8)  (ISA 7.12.2, 16-bit A 16x32 table)
__device__ __forceinline__ v16bf ld_fragA(const u16* row, int c0, int half) {
    union { uint4 q[2]; v16bf v; } u;
    u.q[0] = *(const uint4*)(row + c0 + 8 * half);
    u.q[1] = *(const uint4*)(row + c0 + 16 + 8 * half);
    return u.v;
}

// B-operand fragment (32x16 bf16): lane=column; elems 0..7 at K in [8h,8h+8),
// elems 8..15 at K in [16+8h, 16+8h+8)  (same K order as A fragment)
__device__ __forceinline__ v16bf ld_fragB(const u16* row, int c0, int half) {
    union { uint4 q[2]; v16bf v; } u;
    u.q[0] = *(const uint4*)(row + c0 + 8 * half);
    u.q[1] = *(const uint4*)(row + c0 + 16 + 8 * half);
    return u.v;
}

__device__ __forceinline__ v8f wmma_bf16(v16bf a, v16bf b, v8f c) {
    return __builtin_amdgcn_wmma_f32_16x16x32_bf16(false, a, false, b,
                                                   (short)0, c, false, false);
}

__device__ __forceinline__ v16bf ones_frag() {
    union { u16 s[16]; v16bf v; } u;
#pragma unroll
    for (int i = 0; i < 16; ++i) u.s[i] = 0x3F80;   // bf16 1.0
    return u.v;
}

// ---------------- conversion kernels ----------------

__global__ void cvt_bf16_kernel(const float* __restrict__ in,
                                u16* __restrict__ out, int n) {
    int i = blockIdx.x * blockDim.x + threadIdx.x;
    if (i < n) out[i] = f2bf(in[i]);
}

// W[k][n] fp32 -> Wt[n][k] bf16   (C = 1024)
__global__ void cvtT_kernel(const float* __restrict__ W, u16* __restrict__ Wt) {
    int i = blockIdx.x * blockDim.x + threadIdx.x;   // 0 .. C*C-1, n-major
    int n = i >> 10;
    int k = i & (Cx - 1);
    Wt[(size_t)n * Cx + k] = f2bf(W[(size_t)k * Cx + n]);
}

// ---------------- GEMM: Y = Xbf16 @ W^T + bias ----------------
// per wave: 32(M) x 128(N) tile, K-loop step 32.
// The 8 waves of a block share one N-slice (same B fragments -> WGP$ hits)
// and cover 8 consecutive M-tiles.
// MODE 0: bf16 out as [B,H,T,D]   (Q and K)
// MODE 1: bf16 out as [B,H,D,T]   (V transposed)
// MODE 3: fp32 out as [M,N]       (final projection)
template <int MODE>
__global__ void __launch_bounds__(256)
gemm_kernel(const u16* __restrict__ X, const u16* __restrict__ Wt,
            const float* __restrict__ bias, void* __restrict__ OutP) {
    const int lane = threadIdx.x & 31;
    const int half = lane >> 4, lr = lane & 15;
    const int w    = threadIdx.x >> 5;          // wave in block, 0..7
    const int bid  = blockIdx.x;                // 0..255
    const int nt   = bid & 7;                   // shared N-slice for the block
    const int mt   = (bid >> 3) * 8 + w;        // 0..255, 32 rows each
    const int m0   = mt * 32, n0 = nt * 128;

    const u16* xrow0 = X + (size_t)(m0 + lr) * Cx;
    const u16* xrow1 = X + (size_t)(m0 + 16 + lr) * Cx;

    v8f acc0[8] = {};
    v8f acc1[8] = {};
    for (int k0 = 0; k0 < Cx; k0 += 32) {
        v16bf a0 = ld_fragA(xrow0, k0, half);
        v16bf a1 = ld_fragA(xrow1, k0, half);
        if (k0 + 32 < Cx) {
            __builtin_prefetch(xrow0 + k0 + 32, 0, 3);
            __builtin_prefetch(xrow1 + k0 + 32, 0, 3);
        }
#pragma unroll
        for (int j = 0; j < 8; ++j) {
            v16bf bm = ld_fragB(Wt + (size_t)(n0 + j * 16 + lr) * Cx, k0, half);
            acc0[j] = wmma_bf16(a0, bm, acc0[j]);
            acc1[j] = wmma_bf16(a1, bm, acc1[j]);
        }
    }

#pragma unroll
    for (int j = 0; j < 8; ++j) {
#pragma unroll
        for (int i = 0; i < 8; ++i) {
#pragma unroll
            for (int p = 0; p < 2; ++p) {
                int row = m0 + p * 16 + i + 8 * half;   // C-frag: M = i + 8*half
                int col = n0 + j * 16 + lr;             //         N = lane&15
                float v = (p ? acc1[j][i] : acc0[j][i]) + bias[col];
                if (MODE == 0) {
                    int b = row >> 11, t = row & (Tx - 1);
                    int h = col >> 6,  d = col & (Dx - 1);
                    ((u16*)OutP)[(((size_t)(b * Hx + h) * Tx) + t) * Dx + d] = f2bf(v);
                } else if (MODE == 1) {
                    int b = row >> 11, t = row & (Tx - 1);
                    int h = col >> 6,  d = col & (Dx - 1);
                    ((u16*)OutP)[(((size_t)(b * Hx + h) * Dx) + d) * Tx + t] = f2bf(v);
                } else {
                    ((float*)OutP)[(size_t)row * Cx + col] = v;
                }
            }
        }
    }
}

// ---------------- flash attention ----------------
// one wave = 16 q rows of one head; 8 waves / block (consecutive q-tiles of
// the same head -> K/V fragments shared through WGP$); loop over 32 s-columns
__global__ void __launch_bounds__(256)
attn_kernel(const u16* __restrict__ Q, const u16* __restrict__ K,
            const u16* __restrict__ Vt, u16* __restrict__ A) {
    __shared__ __attribute__((aligned(16))) u16 ldsP[8 * 16 * 32];

    const int lane = threadIdx.x & 31, wavein = threadIdx.x >> 5;
    const int half = lane >> 4, lr = lane & 15;
    const int gw   = blockIdx.x * 8 + wavein;
    const int head = gw >> 7;                    // 128 q-tiles per head
    const int q0   = (gw & 127) * 16;

    const u16* Qh = Q  + (size_t)head * Tx * Dx;
    const u16* Kh = K  + (size_t)head * Tx * Dx;
    const u16* Vh = Vt + (size_t)head * Dx * Tx;
    u16* myP = ldsP + wavein * (16 * 32);

    v16bf qf0 = ld_fragA(Qh + (size_t)(q0 + lr) * Dx, 0, half);
    v16bf qf1 = ld_fragA(Qh + (size_t)(q0 + lr) * Dx, 32, half);
    const v16bf onesB = ones_frag();

    v8f o[4] = {};
    float m[8], l[8];
#pragma unroll
    for (int i = 0; i < 8; ++i) { m[i] = -3.0e38f; l[i] = 0.f; }

    for (int s0 = 0; s0 < Tx; s0 += 32) {
        // scores: two 16x16 tiles, K-dim D=64 => 2 WMMAs each
        v8f sc0 = {}, sc1 = {};
        {
            v16bf kb = ld_fragB(Kh + (size_t)(s0 + lr) * Dx, 0, half);
            sc0 = wmma_bf16(qf0, kb, sc0);
            kb = ld_fragB(Kh + (size_t)(s0 + lr) * Dx, 32, half);
            sc0 = wmma_bf16(qf1, kb, sc0);
            kb = ld_fragB(Kh + (size_t)(s0 + 16 + lr) * Dx, 0, half);
            sc1 = wmma_bf16(qf0, kb, sc1);
            kb = ld_fragB(Kh + (size_t)(s0 + 16 + lr) * Dx, 32, half);
            sc1 = wmma_bf16(qf1, kb, sc1);
        }
        float p0[8], p1[8], alpha[8];
#pragma unroll
        for (int i = 0; i < 8; ++i) {
            p0[i] = sc0[i] * 0.125f;             // 1/sqrt(64)
            p1[i] = sc1[i] * 0.125f;
            // row max across the 16-lane group (stays within half under wave32)
            float t = fmaxf(p0[i], p1[i]);
            t = fmaxf(t, __shfl_xor(t, 1, 32));
            t = fmaxf(t, __shfl_xor(t, 2, 32));
            t = fmaxf(t, __shfl_xor(t, 4, 32));
            t = fmaxf(t, __shfl_xor(t, 8, 32));
            float mn = fmaxf(m[i], t);
            alpha[i] = __expf(m[i] - mn);
            m[i]  = mn;
            p0[i] = __expf(p0[i] - mn);
            p1[i] = __expf(p1[i] - mn);
#pragma unroll
            for (int dt = 0; dt < 4; ++dt) o[dt][i] *= alpha[i];
        }
        // C-layout -> A-layout transpose of P through LDS
#pragma unroll
        for (int i = 0; i < 8; ++i) {
            myP[(i + 8 * half) * 32 + lr]      = f2bf(p0[i]);
            myP[(i + 8 * half) * 32 + 16 + lr] = f2bf(p1[i]);
        }
        __syncthreads();
        v16bf pf = ld_fragA(myP + lr * 32, 0, half);
        // row-sum of P via P @ ones: C element i = sum of row (i+8h),
        // replicated across the lane group -- exactly the layout l[] needs
        v8f sums = {};
        sums = wmma_bf16(pf, onesB, sums);
#pragma unroll
        for (int i = 0; i < 8; ++i) l[i] = l[i] * alpha[i] + sums[i];
#pragma unroll
        for (int dt = 0; dt < 4; ++dt) {
            v16bf vb = ld_fragB(Vh + (size_t)(dt * 16 + lr) * Tx, s0, half);
            o[dt] = wmma_bf16(pf, vb, o[dt]);
        }
        __syncthreads();
    }

    const int bIdx = head / Hx, h = head % Hx;
#pragma unroll
    for (int i = 0; i < 8; ++i) {
        float inv = 1.0f / l[i];
        int t = q0 + i + 8 * half;
#pragma unroll
        for (int dt = 0; dt < 4; ++dt) {
            int d = dt * 16 + lr;
            A[((size_t)(bIdx * Tx + t)) * Cx + h * Dx + d] = f2bf(o[dt][i] * inv);
        }
    }
}

// ---------------- host launch ----------------

extern "C" void kernel_launch(void* const* d_in, const int* in_sizes, int n_in,
                              void* d_out, int out_size, void* d_ws, size_t ws_size,
                              hipStream_t stream) {
    const float* x  = (const float*)d_in[0];
    const float* Wq = (const float*)d_in[1];
    const float* bq = (const float*)d_in[2];
    const float* Wk = (const float*)d_in[3];
    const float* bk = (const float*)d_in[4];
    const float* Wv = (const float*)d_in[5];
    const float* bv = (const float*)d_in[6];
    const float* Wo = (const float*)d_in[7];
    const float* bo = (const float*)d_in[8];

    const size_t SZ_X = (size_t)Mx * Cx;       // 8.4M elems
    const size_t SZ_W = (size_t)Cx * Cx;       // 1M elems

    char* ws = (char*)d_ws;
    size_t off = 0;
    u16* Xb  = (u16*)(ws + off); off += SZ_X * 2;   // also reused as attn output
    u16* Wqb = (u16*)(ws + off); off += SZ_W * 2;
    u16* Wkb = (u16*)(ws + off); off += SZ_W * 2;
    u16* Wvb = (u16*)(ws + off); off += SZ_W * 2;
    u16* Wob = (u16*)(ws + off); off += SZ_W * 2;
    u16* Qb  = (u16*)(ws + off); off += SZ_X * 2;
    u16* Kb  = (u16*)(ws + off); off += SZ_X * 2;
    u16* Vtb = (u16*)(ws + off); off += SZ_X * 2;
    u16* Ab  = Xb;   // x bf16 is dead after QKV GEMMs; reuse for attention out

    cvt_bf16_kernel<<<(int)(SZ_X / 256), 256, 0, stream>>>(x, Xb, (int)SZ_X);
    cvtT_kernel<<<(int)(SZ_W / 256), 256, 0, stream>>>(Wq, Wqb);
    cvtT_kernel<<<(int)(SZ_W / 256), 256, 0, stream>>>(Wk, Wkb);
    cvtT_kernel<<<(int)(SZ_W / 256), 256, 0, stream>>>(Wv, Wvb);
    cvtT_kernel<<<(int)(SZ_W / 256), 256, 0, stream>>>(Wo, Wob);

    // 2048 waves = (8192/32) m-tiles * (1024/128) n-tiles ; 8 waves/block
    gemm_kernel<0><<<256, 256, 0, stream>>>(Xb, Wqb, bq, (void*)Qb);
    gemm_kernel<0><<<256, 256, 0, stream>>>(Xb, Wkb, bk, (void*)Kb);
    gemm_kernel<1><<<256, 256, 0, stream>>>(Xb, Wvb, bv, (void*)Vtb);

    // 64 heads * 128 q-tiles = 8192 waves ; 8 waves/block
    attn_kernel<<<1024, 256, 0, stream>>>(Qb, Kb, Vtb, Ab);

    gemm_kernel<3><<<256, 256, 0, stream>>>(Ab, Wob, bo, d_out);
}
